// Encoder_SigleSeqWithPairBias_34660386079022
// MI455X (gfx1250) — matrix operand order repaired and checked by reading, not compiled, not run on hardware
//
#include <hip/hip_runtime.h>
#include <hip/hip_bf16.h>
#include <math.h>

// ---------------------------------------------------------------------------
// Problem constants (from the reference)
// ---------------------------------------------------------------------------
#define Bn   2
#define Ln   192
#define DSn  384
#define DPn  128
#define Hn   8
#define DHn  32
#define HDn  256            // H*DH
#define ATT_SCALE 0.17677669529663687f   // 1/sqrt(32)

typedef _Float16 v16h  __attribute__((ext_vector_type(16)));
typedef float    v8f   __attribute__((ext_vector_type(8)));
typedef unsigned int u32x4 __attribute__((ext_vector_type(4)));
typedef int      i32x4 __attribute__((ext_vector_type(4)));
typedef int      i32x8 __attribute__((ext_vector_type(8)));

#if defined(__has_builtin)
#if __has_builtin(__builtin_amdgcn_tensor_load_to_lds) && __has_builtin(__builtin_amdgcn_s_wait_tensorcnt)
#define HAVE_TDM 1
#endif
#endif
#ifndef HAVE_TDM
#define HAVE_TDM 0
#endif

__device__ __forceinline__ v8f wmma_f16(v16h a, v16h b, v8f c) {
  // D = A(16x32) * B(32x16) + C, fp32 accumulate
  return __builtin_amdgcn_wmma_f32_16x16x32_f16(
      /*neg_a=*/false, a, /*neg_b=*/false, b,
      /*c_mod=*/(short)0, c, /*reuse_a=*/false, /*reuse_b=*/false);
}

__device__ __forceinline__ float sigmoidf_dev(float x) {
  return 1.0f / (1.0f + __expf(-x));
}

// ---------------------------------------------------------------------------
// TDM: async 2D tile load Global -> LDS (fp32 elements).
// Builds D# Group0/Group1 per CDNA5 ISA 8.3/8.4:
//   g0: count=1 | lds_addr | global_addr[56:0] | type=2
//   g1: data_size=4B, tensor_dim0/1, tile_dim0/1, tensor_dim0_stride
// tile_w elements per row (contiguous), tile_h rows, row_stride in elements.
// ---------------------------------------------------------------------------
__device__ __forceinline__ void tdm_load_2d(void* lds_ptr, const float* gptr,
                                            unsigned tile_w, unsigned tile_h,
                                            unsigned long long row_stride) {
#if HAVE_TDM
  unsigned long long ga = (unsigned long long)(uintptr_t)gptr;
  unsigned lds_off = (unsigned)(uintptr_t)lds_ptr;   // low 32 bits = LDS offset
  u32x4 g0;
  g0[0] = 1u;                                        // count=1, user mode
  g0[1] = lds_off;
  g0[2] = (unsigned)(ga & 0xffffffffu);
  g0[3] = (unsigned)((ga >> 32) & 0x01ffffffu) | (2u << 30);   // type=2 ("image")
  unsigned long long sb = row_stride;                // in data_size units
  i32x8 g1;
  g1[0] = (int)(2u << 16);                           // data_size=4B, wg_mask=0
  g1[1] = (int)((tile_w & 0xffffu) << 16);           // tensor_dim0[15:0] @ 63:48
  g1[2] = (int)((tile_w >> 16) & 0xffffu) |          // tensor_dim0[31:16]
          (int)((tile_h & 0xffffu) << 16);           // tensor_dim1[15:0]
  g1[3] = (int)((tile_h >> 16) & 0xffffu) |          // tensor_dim1[31:16]
          (int)((tile_w & 0xffffu) << 16);           // tile_dim0
  g1[4] = (int)(tile_h & 0xffffu);                   // tile_dim1 (tile_dim2=0)
  g1[5] = (int)(sb & 0xffffffffull);                 // stride[31:0]
  g1[6] = (int)((sb >> 32) & 0xffffull);             // stride[47:32]
  g1[7] = 0;
  i32x4 z4 = {0, 0, 0, 0};
#if __clang_major__ >= 23
  i32x8 z8 = {0, 0, 0, 0, 0, 0, 0, 0};
  __builtin_amdgcn_tensor_load_to_lds(g0, g1, z4, z4, z8, 0);
#else
  __builtin_amdgcn_tensor_load_to_lds(g0, g1, z4, z4, 0);
#endif
#else
  // fallback: manual wave staging (one wave executes this)
  int lane = threadIdx.x & 31;
  float* dst = (float*)lds_ptr;
  for (unsigned r = 0; r < tile_h; ++r)
    for (unsigned c = lane; c < tile_w; c += 32)
      dst[(size_t)r * tile_w + c] = gptr[(size_t)r * row_stride + c];
#endif
}

__device__ __forceinline__ void tdm_wait() {
#if HAVE_TDM
  __builtin_amdgcn_s_wait_tensorcnt(0);
#endif
}

// ---------------------------------------------------------------------------
// LayerNorm: one block per row
// ---------------------------------------------------------------------------
__global__ void ln_kernel(const float* __restrict__ x, const float* __restrict__ g,
                          const float* __restrict__ b, float* __restrict__ y, int D) {
  int row = blockIdx.x;
  const float* xr = x + (size_t)row * D;
  float* yr = y + (size_t)row * D;
  float s = 0.f, s2 = 0.f;
  for (int i = threadIdx.x; i < D; i += blockDim.x) {
    float v = xr[i]; s += v; s2 += v * v;
  }
  __shared__ float r0[128], r1[128];
  r0[threadIdx.x] = s; r1[threadIdx.x] = s2;
  __syncthreads();
  for (int off = 64; off > 0; off >>= 1) {
    if (threadIdx.x < off) { r0[threadIdx.x] += r0[threadIdx.x + off];
                             r1[threadIdx.x] += r1[threadIdx.x + off]; }
    __syncthreads();
  }
  float mean = r0[0] / (float)D;
  float var  = r1[0] / (float)D - mean * mean;
  float inv  = rsqrtf(var + 1e-5f);
  for (int i = threadIdx.x; i < D; i += blockDim.x)
    yr[i] = (xr[i] - mean) * inv * g[i] + b[i];
}

// ---------------------------------------------------------------------------
// Generic GEMM: Y[M,N] = act(X[M,K] @ W[K,N] + bias)
// Requirements (all call sites satisfy): M % 64 == 0, K % 32 == 0.
// 64x64 tile per block, 4 waves. X tile arrives via TDM (async tensor DMA) as
// fp32 in LDS; W tile staged transposed as fp16 by the VALU path.
// ---------------------------------------------------------------------------
__launch_bounds__(128)
__global__ void gemm_kernel(const float* __restrict__ X, const float* __restrict__ W,
                            const float* __restrict__ bias, float* __restrict__ Y,
                            int M, int N, int K, int act) {
  __shared__ float    Asf[64][32];  // [m][k]  fp32, TDM destination
  __shared__ _Float16 Bs[64][32];   // [n][k]  fp16, W tile transposed
  const int m0 = blockIdx.y * 64;
  const int n0 = blockIdx.x * 64;
  const int tid  = threadIdx.x;
  const int wave = tid >> 5;
  const int lane = tid & 31;
  const int ml = lane & 15;
  const int ko = (lane < 16) ? 0 : 8;    // unused; A frag also uses kb
  const int kb = (lane < 16) ? 0 : 16;   // B-frag K base

  v8f acc[4] = {};

  for (int k0 = 0; k0 < K; k0 += 32) {
    // async tensor DMA: 64x32 fp32 X tile -> LDS (issued by wave 0 only)
    if (wave == 0)
      tdm_load_2d(&Asf[0][0], X + (size_t)m0 * K + k0, 32, 64, (unsigned long long)K);
    // stage W tile transposed to fp16 (clamped column index: out-of-range
    // columns produce garbage that is never stored)
    for (int i = tid; i < 64 * 32; i += 128) {
      int n = i >> 5, c = i & 31;
      int gn = n0 + n; if (gn >= N) gn = N - 1;
      Bs[n][c] = (_Float16)W[(size_t)(k0 + c) * N + gn];
    }
    if (k0 + 32 < K && wave == 0)
      __builtin_prefetch(&W[(size_t)(k0 + 32 + lane) * N + n0], 0, 1);
    if (wave == 0) tdm_wait();
    __syncthreads();

    // A fragment: convert the wave's 16 rows x 32 K from fp32 LDS
    v16h a;
    const float* ar = &Asf[wave * 16 + ml][0];
#pragma unroll
    for (int e = 0; e < 8; ++e) {
      a[e]     = (_Float16)ar[kb + e];
      a[8 + e] = (_Float16)ar[kb + 8 + e];
    }
#pragma unroll
    for (int j = 0; j < 4; ++j) {
      v16h bf;
      const _Float16* br = &Bs[j * 16 + ml][0];
#pragma unroll
      for (int e = 0; e < 16; ++e) bf[e] = br[kb + e];
      acc[j] = wmma_f16(a, bf, acc[j]);
    }
    __syncthreads();
  }

  // epilogue (M % 64 == 0: no row guard)
  const int nl = lane & 15;
  const int mb = (lane < 16) ? 0 : 8;
#pragma unroll
  for (int j = 0; j < 4; ++j) {
    int gn = n0 + j * 16 + nl;
    if (gn >= N) continue;
    float bv = bias ? bias[gn] : 0.f;
#pragma unroll
    for (int r = 0; r < 8; ++r) {
      int gm = m0 + wave * 16 + mb + r;
      float v = acc[j][r] + bv;
      if (act == 1) v = fmaxf(v, 0.f);
      Y[(size_t)gm * N + gn] = v;
    }
  }
}

// ---------------------------------------------------------------------------
// Fused multi-head attention over T=192 keys, DH=32, one wave per block.
// Element (n,t,h,d) of Q/K/V/O/gate lives at  base + n*sN + t*sT + h*32 + d.
// bias (optional) addressed as  bias + q*bsQ + k*bsK + h.
// gate (optional): O *= sigmoid(gate) on the epilogue.
// K and V tiles (192x32 fp32 each) are fetched by the Tensor Data Mover.
// grid = (L/16 query tiles, H, N)
// ---------------------------------------------------------------------------
__launch_bounds__(32)
__global__ void attn_kernel(const float* __restrict__ Q, const float* __restrict__ K,
                            const float* __restrict__ V, float* __restrict__ O,
                            const float* __restrict__ gate, const float* __restrict__ bias,
                            long long sN, long long sT, long long bsQ, long long bsK,
                            float scale) {
  __shared__ float Ks[Ln][32];   // 24 KB
  __shared__ float Vs[Ln][32];   // 24 KB
  __shared__ float sc[16][Ln];   // 12 KB scores/probabilities
  const int tile = blockIdx.x;
  const int h    = blockIdx.y;
  const int n    = blockIdx.z;
  const int lane = threadIdx.x;
  const int ml = lane & 15;
  const int ko = (lane < 16) ? 0 : 8;
  const int kb = (lane < 16) ? 0 : 16;
  const size_t headOff = (size_t)n * sN + (size_t)h * 32;

  // async tensor DMA: whole K and V head-slices into LDS (in-order, one wait)
  tdm_load_2d(&Ks[0][0], K + headOff, 32, Ln, (unsigned long long)sT);
  tdm_load_2d(&Vs[0][0], V + headOff, 32, Ln, (unsigned long long)sT);

  // A fragment: 16 queries x 32 dims (scaled), from global
  v16h aq;
  {
    const float* qrow = Q + headOff + (size_t)(tile * 16 + ml) * sT;
#pragma unroll
    for (int e = 0; e < 8; ++e) {
      aq[e]     = (_Float16)(qrow[kb + e] * scale);
      aq[8 + e] = (_Float16)(qrow[kb + 8 + e] * scale);
    }
  }
  tdm_wait();
  __syncthreads();

  // logits = Q K^T (+bias) -> LDS
  for (int c = 0; c < Ln / 16; ++c) {
    v16h bk;
    const float* krow = &Ks[c * 16 + ml][0];
#pragma unroll
    for (int e = 0; e < 16; ++e) bk[e] = (_Float16)krow[kb + e];
    v8f cc = {};
    cc = wmma_f16(aq, bk, cc);
    const int nl = lane & 15;
    const int mb = (lane < 16) ? 0 : 8;
#pragma unroll
    for (int r = 0; r < 8; ++r) {
      int m   = mb + r;
      int key = c * 16 + nl;
      float v = cc[r];
      if (bias) v += bias[(size_t)(tile * 16 + m) * bsQ + (size_t)key * bsK + h];
      sc[m][key] = v;
    }
  }
  __syncthreads();

  // softmax: lanes 0..15 each own one row
  if (lane < 16) {
    float mx = -1e30f;
    for (int j = 0; j < Ln; ++j) mx = fmaxf(mx, sc[lane][j]);
    float sum = 0.f;
    for (int j = 0; j < Ln; ++j) { float e = __expf(sc[lane][j] - mx); sc[lane][j] = e; sum += e; }
    float inv = 1.f / sum;
    for (int j = 0; j < Ln; ++j) sc[lane][j] *= inv;
  }
  __syncthreads();

  // O = P @ V  (contract over 192 keys in chunks of 32)
  v8f o0 = {}, o1 = {};
  for (int c = 0; c < Ln / 32; ++c) {
    v16h pa;
#pragma unroll
    for (int e = 0; e < 8; ++e) {
      pa[e]     = (_Float16)sc[ml][c * 32 + kb + e];
      pa[8 + e] = (_Float16)sc[ml][c * 32 + kb + 8 + e];
    }
#pragma unroll
    for (int j = 0; j < 2; ++j) {
      v16h vb;
      int dcol = j * 16 + ml;
#pragma unroll
      for (int e = 0; e < 16; ++e)
        vb[e] = (_Float16)Vs[c * 32 + kb + e][dcol];
      if (j == 0) o0 = wmma_f16(pa, vb, o0);
      else        o1 = wmma_f16(pa, vb, o1);
    }
  }

  // epilogue with optional sigmoid gating
  const int nl = lane & 15;
  const int mb = (lane < 16) ? 0 : 8;
#pragma unroll
  for (int r = 0; r < 8; ++r) {
    int tq = tile * 16 + mb + r;
    size_t rowoff = headOff + (size_t)tq * sT;
#pragma unroll
    for (int j = 0; j < 2; ++j) {
      int d = j * 16 + nl;
      float v = (j == 0) ? o0[r] : o1[r];
      if (gate) v *= sigmoidf_dev(gate[rowoff + d]);
      O[rowoff + d] = v;
    }
  }
}

// ---------------------------------------------------------------------------
// Triangle-update einsum, per channel c (32 channels):
//   outgoing: o[i,j,c] = sum_k A[i,k,c] * B[j,k,c]
//   incoming: o[i,j,c] = sum_k A[k,i,c] * B[k,j,c]
// selected via strides.  grid = (L/16, L/16, 32), one wave per block.
// ---------------------------------------------------------------------------
__launch_bounds__(32)
__global__ void trimul_kernel(const float* __restrict__ A, const float* __restrict__ Bm,
                              float* __restrict__ Oo,
                              long long sAi, long long sAk, long long sBj, long long sBk) {
  const int jt = blockIdx.x, it = blockIdx.y, c = blockIdx.z;
  const int lane = threadIdx.x;
  const int ml = lane & 15;
  const int ko = (lane < 16) ? 0 : 8;
  const int kb = (lane < 16) ? 0 : 16;
  const int i = it * 16 + ml;
  const int j = jt * 16 + ml;
  v8f acc = {};
  for (int kc = 0; kc < Ln / 32; ++kc) {
    v16h a, b;
#pragma unroll
    for (int e = 0; e < 8; ++e) {
      a[e]     = (_Float16)A[(size_t)i * sAi + (size_t)(kc * 32 + kb + e) * sAk + c];
      a[8 + e] = (_Float16)A[(size_t)i * sAi + (size_t)(kc * 32 + kb + 8 + e) * sAk + c];
    }
#pragma unroll
    for (int e = 0; e < 16; ++e)
      b[e] = (_Float16)Bm[(size_t)j * sBj + (size_t)(kc * 32 + kb + e) * sBk + c];
    acc = wmma_f16(a, b, acc);
  }
  const int nl = lane & 15;
  const int mb = (lane < 16) ? 0 : 8;
#pragma unroll
  for (int r = 0; r < 8; ++r) {
    int gi = it * 16 + mb + r;
    int gj = jt * 16 + nl;
    Oo[((size_t)gi * Ln + gj) * DHn + c] = acc[r];
  }
}

// ---------------------------------------------------------------------------
// outer[i,j,m,n] = (a[0,i,m]*b[0,j,n] + a[1,i,m]*b[1,j,n]) / B   (float4 wide)
// ---------------------------------------------------------------------------
__global__ void outer_kernel(const float* __restrict__ av, const float4* __restrict__ bv4,
                             float4* __restrict__ out, int total4) {
  int idx = blockIdx.x * blockDim.x + threadIdx.x;
  if (idx >= total4) return;
  int linear = idx << 2;
  int mn = linear & 1023;
  int pr = linear >> 10;
  int m = mn >> 5, n0 = mn & 31;          // n0 multiple of 4
  int i = pr / Ln, j = pr % Ln;
  float a0 = av[(size_t)i * 32 + m];
  float a1 = av[(size_t)(Ln + i) * 32 + m];
  float4 b0 = bv4[((size_t)j * 32 + n0) >> 2];
  float4 b1 = bv4[((size_t)(Ln + j) * 32 + n0) >> 2];
  float4 r;
  r.x = 0.5f * (a0 * b0.x + a1 * b1.x);
  r.y = 0.5f * (a0 * b0.y + a1 * b1.y);
  r.z = 0.5f * (a0 * b0.z + a1 * b1.z);
  r.w = 0.5f * (a0 * b0.w + a1 * b1.w);
  out[idx] = r;
}

// ---------------------------------------------------------------------------
// elementwise helpers (float4 wide; all sizes divisible by 4)
// ---------------------------------------------------------------------------
__global__ void add2_kernel(const float4* __restrict__ a, const float4* __restrict__ b,
                            float4* __restrict__ y, int n4) {
  int i = blockIdx.x * blockDim.x + threadIdx.x;
  if (i < n4) {
    float4 u = a[i], v = b[i];
    y[i] = make_float4(u.x + v.x, u.y + v.y, u.z + v.z, u.w + v.w);
  }
}
__global__ void addin_kernel(float4* __restrict__ y, const float4* __restrict__ x, int n4) {
  int i = blockIdx.x * blockDim.x + threadIdx.x;
  if (i < n4) {
    float4 u = y[i], v = x[i];
    y[i] = make_float4(u.x + v.x, u.y + v.y, u.z + v.z, u.w + v.w);
  }
}
__global__ void gatemul_kernel(float4* __restrict__ dst, const float4* __restrict__ g,
                               const float4* __restrict__ x, int n4) {
  int i = blockIdx.x * blockDim.x + threadIdx.x;
  if (i < n4) {
    float4 gv = g[i], xv = x[i];
    dst[i] = make_float4(sigmoidf_dev(gv.x) * xv.x, sigmoidf_dev(gv.y) * xv.y,
                         sigmoidf_dev(gv.z) * xv.z, sigmoidf_dev(gv.w) * xv.w);
  }
}
__global__ void resgate_kernel(float4* __restrict__ z, const float4* __restrict__ g,
                               const float4* __restrict__ x, int n4) {
  int i = blockIdx.x * blockDim.x + threadIdx.x;
  if (i < n4) {
    float4 zv = z[i], gv = g[i], xv = x[i];
    z[i] = make_float4(zv.x + sigmoidf_dev(gv.x) * xv.x, zv.y + sigmoidf_dev(gv.y) * xv.y,
                       zv.z + sigmoidf_dev(gv.z) * xv.z, zv.w + sigmoidf_dev(gv.w) * xv.w);
  }
}

// ---------------------------------------------------------------------------
// Parameter indices (setup_inputs dict order, params flattened recursively)
// ---------------------------------------------------------------------------
enum {
  IN_SEQ = 0, IN_PAIR,
  SAPB_LNS_G, SAPB_LNS_B, SAPB_LNP_G, SAPB_LNP_B,
  SAPB_Q_W, SAPB_Q_B, SAPB_K_W, SAPB_K_B, SAPB_V_W, SAPB_V_B,
  SAPB_BIAS_W, SAPB_BIAS_B, SAPB_GATE_W, SAPB_GATE_B, SAPB_O_W, SAPB_O_B,
  SA_LN_G, SA_LN_B, SA_Q_W, SA_Q_B, SA_K_W, SA_K_B, SA_V_W, SA_V_B, SA_O_W, SA_O_B,
  ST_LN_G, ST_LN_B, ST_L1_W, ST_L1_B, ST_L2_W, ST_L2_B,
  TUO_LN_G, TUO_LN_B, TUO_A_W, TUO_A_B, TUO_AG_W, TUO_AG_B, TUO_B_W, TUO_B_B,
  TUO_BG_W, TUO_BG_B, TUO_LNO_G, TUO_LNO_B, TUO_G_W, TUO_G_B, TUO_O_W, TUO_O_B,
  TUI_LN_G, TUI_LN_B, TUI_A_W, TUI_A_B, TUI_AG_W, TUI_AG_B, TUI_B_W, TUI_B_B,
  TUI_BG_W, TUI_BG_B, TUI_LNO_G, TUI_LNO_B, TUI_G_W, TUI_G_B, TUI_O_W, TUI_O_B,
  TAS_LN_G, TAS_LN_B, TAS_Q_W, TAS_Q_B, TAS_K_W, TAS_K_B, TAS_V_W, TAS_V_B,
  TAS_BIAS_W, TAS_BIAS_B, TAS_GATE_W, TAS_GATE_B, TAS_O_W, TAS_O_B,
  TAE_LN_G, TAE_LN_B, TAE_Q_W, TAE_Q_B, TAE_K_W, TAE_K_B, TAE_V_W, TAE_V_B,
  TAE_BIAS_W, TAE_BIAS_B, TAE_GATE_W, TAE_GATE_B, TAE_O_W, TAE_O_B,
  PT_LN_G, PT_LN_B, PT_L1_W, PT_L1_B, PT_L2_W, PT_L2_B,
  P_LN_G, P_LN_B, P_L1_W, P_L1_B, P_L2_W, P_L2_B, P_L_W, P_L_B
};

static void launch_gemm(const float* X, const float* W, const float* bias, float* Y,
                        int M, int N, int K, int act, hipStream_t s) {
  dim3 g((N + 63) / 64, (M + 63) / 64);
  gemm_kernel<<<g, 128, 0, s>>>(X, W, bias, Y, M, N, K, act);
}

// ---------------------------------------------------------------------------
extern "C" void kernel_launch(void* const* d_in, const int* in_sizes, int n_in,
                              void* d_out, int out_size, void* d_ws, size_t ws_size,
                              hipStream_t stream) {
  (void)in_sizes; (void)n_in; (void)out_size; (void)ws_size;
  #define PF(i) ((const float*)d_in[(i)])
  #define F4(p) ((float4*)(p))
  #define CF4(p) ((const float4*)(p))

  const float* seq  = PF(IN_SEQ);    // [B,L,DS]
  const float* pair = PF(IN_PAIR);   // [1,L,L,DP]

  // sizes (floats)
  const size_t S    = (size_t)Bn * Ln * DSn;        // 147456
  const size_t QKV  = (size_t)Bn * Ln * HDn;        // 98304
  const size_t ZB   = (size_t)Ln * Ln * DPn;        // 4718592
  const size_t Z32  = (size_t)Ln * Ln * DHn;        // 1179648
  const size_t Z256 = (size_t)Ln * Ln * HDn;        // 9437184
  const size_t Z8   = (size_t)Ln * Ln * Hn;         // 294912
  const size_t H1   = (size_t)Bn * Ln * 4 * DSn;    // 589824
  const size_t AB   = (size_t)Bn * Ln * DHn;        // 12288
  const size_t OUT1 = (size_t)Ln * Ln * DHn * DHn;  // 37748736

  // outputs accumulate in place in d_out: s first, then z
  float* s_cur = (float*)d_out;
  float* z_cur = (float*)d_out + S;

  // workspace carve-up
  float* ws = (float*)d_ws;
  size_t off = 0;
  auto alloc = [&](size_t n) { float* p = ws + off; off += (n + 63) & ~(size_t)63; return p; };
  float* xs     = alloc(S);      // LN of seq-stream
  float* stmp   = alloc(S);
  float* qb     = alloc(QKV);
  float* kb2    = alloc(QKV);
  float* vb     = alloc(QKV);
  float* gb     = alloc(QKV);
  float* ob     = alloc(QKV);
  float* h1b    = alloc(H1);
  float* biasHb = alloc(Z8);
  float* zlnb   = alloc(ZB);     // LN of pair-stream (reused each z-stage)
  float* avec   = alloc(AB);
  float* bvec   = alloc(AB);
  float* ta     = alloc(Z32);
  float* tb_    = alloc(Z32);
  float* tmp32  = alloc(Z32);
  float* to_    = alloc(Z32);
  float* tln    = alloc(Z32);
  float* tg     = alloc(ZB);
  float* tproj  = alloc(ZB);
  float* oz     = alloc(Z256);
  float* outerB = alloc(OUT1);   // reused: qz/kz/vz/gz (4*Z256 == OUT1), later pth
  float* qz = outerB;
  float* kz = outerB + Z256;
  float* vz = outerB + 2 * Z256;
  float* gz = outerB + 3 * Z256;
  float* pth = outerB;           // [L*L, 512] pair-transition hidden

  const int BL = Bn * Ln;        // 384 seq rows   (M % 64 == 0)
  const int LL = Ln * Ln;        // 36864 pair rows (M % 64 == 0)

  // ============ Stage A: seq attention with pair bias ============
  ln_kernel<<<BL, 128, 0, stream>>>(seq, PF(SAPB_LNS_G), PF(SAPB_LNS_B), xs, DSn);
  launch_gemm(xs, PF(SAPB_Q_W),    PF(SAPB_Q_B),    qb,  BL, HDn, DSn, 0, stream);
  launch_gemm(xs, PF(SAPB_K_W),    PF(SAPB_K_B),    kb2, BL, HDn, DSn, 0, stream);
  launch_gemm(xs, PF(SAPB_V_W),    PF(SAPB_V_B),    vb,  BL, HDn, DSn, 0, stream);
  launch_gemm(xs, PF(SAPB_GATE_W), PF(SAPB_GATE_B), gb,  BL, HDn, DSn, 0, stream);
  ln_kernel<<<LL, 128, 0, stream>>>(pair, PF(SAPB_LNP_G), PF(SAPB_LNP_B), zlnb, DPn);
  launch_gemm(zlnb, PF(SAPB_BIAS_W), PF(SAPB_BIAS_B), biasHb, LL, Hn, DPn, 0, stream);
  // bias_t[h,i,j] = biasH[i,j,h]  -> bsQ = L*H, bsK = H
  attn_kernel<<<dim3(Ln / 16, Hn, Bn), 32, 0, stream>>>(
      qb, kb2, vb, ob, gb, biasHb,
      (long long)Ln * HDn, HDn, (long long)Ln * Hn, Hn, ATT_SCALE);
  launch_gemm(ob, PF(SAPB_O_W), PF(SAPB_O_B), stmp, BL, DSn, HDn, 0, stream);
  add2_kernel<<<((int)(S / 4) + 255) / 256, 256, 0, stream>>>(CF4(seq), CF4(stmp), F4(s_cur), (int)(S / 4));

  // ============ Stage B: plain seq attention ============
  ln_kernel<<<BL, 128, 0, stream>>>(s_cur, PF(SA_LN_G), PF(SA_LN_B), xs, DSn);
  launch_gemm(xs, PF(SA_Q_W), PF(SA_Q_B), qb,  BL, HDn, DSn, 0, stream);
  launch_gemm(xs, PF(SA_K_W), PF(SA_K_B), kb2, BL, HDn, DSn, 0, stream);
  launch_gemm(xs, PF(SA_V_W), PF(SA_V_B), vb,  BL, HDn, DSn, 0, stream);
  attn_kernel<<<dim3(Ln / 16, Hn, Bn), 32, 0, stream>>>(
      qb, kb2, vb, ob, nullptr, nullptr,
      (long long)Ln * HDn, HDn, 0, 0, ATT_SCALE);
  launch_gemm(ob, PF(SA_O_W), PF(SA_O_B), stmp, BL, DSn, HDn, 0, stream);
  addin_kernel<<<((int)(S / 4) + 255) / 256, 256, 0, stream>>>(F4(s_cur), CF4(stmp), (int)(S / 4));

  // ============ Stage C: seq transition ============
  ln_kernel<<<BL, 128, 0, stream>>>(s_cur, PF(ST_LN_G), PF(ST_LN_B), xs, DSn);
  launch_gemm(xs,  PF(ST_L1_W), PF(ST_L1_B), h1b,  BL, 4 * DSn, DSn, 1, stream);
  launch_gemm(h1b, PF(ST_L2_W), PF(ST_L2_B), stmp, BL, DSn, 4 * DSn, 0, stream);
  addin_kernel<<<((int)(S / 4) + 255) / 256, 256, 0, stream>>>(F4(s_cur), CF4(stmp), (int)(S / 4));

  // ============ Stage D: outer-product pair bias ============
  ln_kernel<<<BL, 128, 0, stream>>>(s_cur, PF(P_LN_G), PF(P_LN_B), xs, DSn);
  launch_gemm(xs, PF(P_L1_W), PF(P_L1_B), avec, BL, DHn, DSn, 0, stream);
  launch_gemm(xs, PF(P_L2_W), PF(P_L2_B), bvec, BL, DHn, DSn, 0, stream);
  outer_kernel<<<((int)(OUT1 / 4) + 255) / 256, 256, 0, stream>>>(avec, CF4(bvec), F4(outerB), (int)(OUT1 / 4));
  launch_gemm(outerB, PF(P_L_W), PF(P_L_B), tproj, LL, DPn, DHn * DHn, 0, stream);
  add2_kernel<<<((int)(ZB / 4) + 255) / 256, 256, 0, stream>>>(CF4(pair), CF4(tproj), F4(z_cur), (int)(ZB / 4));

  // ============ Stages E/F: triangle updates (outgoing, incoming) ============
  for (int pass = 0; pass < 2; ++pass) {
    const int base = (pass == 0) ? TUO_LN_G : TUI_LN_G;
    ln_kernel<<<LL, 128, 0, stream>>>(z_cur, PF(base + 0), PF(base + 1), zlnb, DPn);
    launch_gemm(zlnb, PF(base + 2), PF(base + 3), ta,    LL, DHn, DPn, 0, stream);  // a
    launch_gemm(zlnb, PF(base + 4), PF(base + 5), tmp32, LL, DHn, DPn, 0, stream);  // ag
    gatemul_kernel<<<((int)(Z32 / 4) + 255) / 256, 256, 0, stream>>>(F4(ta), CF4(tmp32), CF4(ta), (int)(Z32 / 4));
    launch_gemm(zlnb, PF(base + 6), PF(base + 7), tb_,   LL, DHn, DPn, 0, stream);  // b
    launch_gemm(zlnb, PF(base + 8), PF(base + 9), tmp32, LL, DHn, DPn, 0, stream);  // bg
    gatemul_kernel<<<((int)(Z32 / 4) + 255) / 256, 256, 0, stream>>>(F4(tb_), CF4(tmp32), CF4(tb_), (int)(Z32 / 4));
    const long long big = (long long)Ln * DHn, sml = DHn;
    if (pass == 0)  // outgoing: o[i,j,c] = sum_k a[i,k,c]*b[j,k,c]
      trimul_kernel<<<dim3(Ln / 16, Ln / 16, DHn), 32, 0, stream>>>(
          ta, tb_, to_, big, sml, big, sml);
    else            // incoming: o[i,j,c] = sum_k a[k,i,c]*b[k,j,c]
      trimul_kernel<<<dim3(Ln / 16, Ln / 16, DHn), 32, 0, stream>>>(
          ta, tb_, to_, sml, big, sml, big);
    ln_kernel<<<LL, 128, 0, stream>>>(to_, PF(base + 10), PF(base + 11), tln, DHn);
    launch_gemm(tln,  PF(base + 14), PF(base + 15), tproj, LL, DPn, DHn, 0, stream); // o
    launch_gemm(zlnb, PF(base + 12), PF(base + 13), tg,    LL, DPn, DPn, 0, stream); // g
    resgate_kernel<<<((int)(ZB / 4) + 255) / 256, 256, 0, stream>>>(F4(z_cur), CF4(tg), CF4(tproj), (int)(ZB / 4));
  }

  // ============ Stages G/H: triangle attention (start, end) ============
  for (int pass = 0; pass < 2; ++pass) {
    const int base = (pass == 0) ? TAS_LN_G : TAE_LN_G;
    ln_kernel<<<LL, 128, 0, stream>>>(z_cur, PF(base + 0), PF(base + 1), zlnb, DPn);
    launch_gemm(zlnb, PF(base + 2),  PF(base + 3),  qz, LL, HDn, DPn, 0, stream);
    launch_gemm(zlnb, PF(base + 4),  PF(base + 5),  kz, LL, HDn, DPn, 0, stream);
    launch_gemm(zlnb, PF(base + 6),  PF(base + 7),  vz, LL, HDn, DPn, 0, stream);
    launch_gemm(zlnb, PF(base + 8),  PF(base + 9),  biasHb, LL, Hn, DPn, 0, stream);
    launch_gemm(zlnb, PF(base + 10), PF(base + 11), gz, LL, HDn, DPn, 0, stream);
    if (pass == 0) {
      // start: per row i, attend over j/k; bias bt[h,j,k]=bias[j,k,h]
      attn_kernel<<<dim3(Ln / 16, Hn, Ln), 32, 0, stream>>>(
          qz, kz, vz, oz, gz, biasHb,
          (long long)Ln * HDn, HDn, (long long)Ln * Hn, Hn, ATT_SCALE);
    } else {
      // end: per column j, attend over i/k; bias bt[h,i,k]=bias[k,i,h]
      attn_kernel<<<dim3(Ln / 16, Hn, Ln), 32, 0, stream>>>(
          qz, kz, vz, oz, gz, biasHb,
          HDn, (long long)Ln * HDn, Hn, (long long)Ln * Hn, ATT_SCALE);
    }
    launch_gemm(oz, PF(base + 12), PF(base + 13), tproj, LL, DPn, HDn, 0, stream);
    addin_kernel<<<((int)(ZB / 4) + 255) / 256, 256, 0, stream>>>(F4(z_cur), CF4(tproj), (int)(ZB / 4));
  }

  // ============ Stage I: pair transition ============
  ln_kernel<<<LL, 128, 0, stream>>>(z_cur, PF(PT_LN_G), PF(PT_LN_B), zlnb, DPn);
  launch_gemm(zlnb, PF(PT_L1_W), PF(PT_L1_B), pth,   LL, 4 * DPn, DPn, 1, stream);
  launch_gemm(pth,  PF(PT_L2_W), PF(PT_L2_B), tproj, LL, DPn, 4 * DPn, 0, stream);
  addin_kernel<<<((int)(ZB / 4) + 255) / 256, 256, 0, stream>>>(F4(z_cur), CF4(tproj), (int)(ZB / 4));
  // s_cur / z_cur already live in d_out — done.
  #undef PF
  #undef F4
  #undef CF4
}
